// Encoder_23235773071575
// MI455X (gfx1250) — matrix operand order repaired and checked by reading, not compiled, not run on hardware
//
#include <hip/hip_runtime.h>
#include <hip/hip_bf16.h>
#include <math.h>

// ---------------------------------------------------------------------------
// DA-RNN input-attention encoder, MI455X (gfx1250, wave32, WMMA).
//
//  * Step-invariant precompute: Xw1 = inp_t @ w1[:,512:]^T (bf16),
//    P2 = Xw1 @ w2^T (f32, 25MB -> L2 resident), W12 = w2 @ w1[:,:512] (bf16),
//    all weights pre-converted to bf16 once so inner loops do native bf16
//    b128/b32 loads with zero per-element cvt.
//  * Per step (192 sequential, launch-bound -> 3 kernels/step):
//      S1 k_hc:   hc2 = [h|c] @ W12^T + (w2@b1+b2)           (512x512x192 WMMA)
//      S2 k_attn: logit = b3 + w3·tanh(hc2+P2); softmax; xw  (fused, LDS-staged)
//      S3 k_lstm: bidirectional LSTM gates via WMMA, 4 gate accumulators per
//                 wave -> in-register state update; h kept bf16-only (GEMM A),
//                 c kept f32 (+bf16 copy for S1); ping-pong buffers.
//  All matrix math on v_wmma_f32_16x16x32_bf16 (K=32/instr; the f32 WMMA is
//  K=4/instr -> bf16 is the CDNA5 throughput path).
// ---------------------------------------------------------------------------

#define kB 256
#define kT 192
#define kD 128
#define kH 256
#define kN1 192           // attention hidden dim (== T)
#define kKA 704           // 2H + T
#define kBD (kB * kD)     // 32768 attention rows
#define kST (2 * kB * kH) // 131072 state elements per (h or c)

typedef __attribute__((ext_vector_type(16))) __bf16 v16bf;
typedef __attribute__((ext_vector_type(8)))  __bf16 v8bf;
typedef __attribute__((ext_vector_type(2)))  __bf16 v2bf;
typedef __attribute__((ext_vector_type(8)))  float  v8f;

__device__ inline v8f wmma_bf16(v16bf a, v16bf b, v8f c) {
  return __builtin_amdgcn_wmma_f32_16x16x32_bf16(false, a, false, b, (short)0, c,
                                                 false, false);
}

// A fragment (16x32 bf16), ISA 7.12.2 wave32 layout, f32 source (gather path):
// row m = lane&15 ; lane group g = lane>>4 ; pair p -> K = (p<4?2p:16+2(p-4))+8g.
template <typename F>
__device__ inline v16bf load_a_frag_f32(F f, int lane) {
  const int r = lane & 15, g = lane >> 4;
  v16bf a;
#pragma unroll
  for (int e = 0; e < 16; ++e) {
    const int p = e >> 1, w = e & 1;
    const int k = ((p < 4) ? (p * 2) : (16 + (p - 4) * 2)) + g * 8 + w;
    a[e] = (__bf16)f(r, k);
  }
  return a;
}

// A fragment from bf16 source: pairs (k even, k+1) are 4B-aligned dword loads.
template <typename F>
__device__ inline v16bf load_a_frag_bf(F f2 /*(r, k_even) -> v2bf*/, int lane) {
  const int r = lane & 15, g = lane >> 4;
  v16bf a;
#pragma unroll
  for (int p = 0; p < 8; ++p) {
    const int k = ((p < 4) ? (p * 2) : (16 + (p - 4) * 2)) + g * 8;
    const v2bf pr = f2(r, k);
    a[2 * p] = pr[0];
    a[2 * p + 1] = pr[1];
  }
  return a;
}

// B fragment (32x16 bf16): col n = lane&15; each lane reads K 8g..8g+7 and
// 16+8g..16+8g+7 = two 16B vector loads (rows are 16B aligned).
__device__ inline v16bf load_b_frag_bf(const __bf16* __restrict__ W, int ldw,
                                       int n0, int k0, int lane) {
  const int n = n0 + (lane & 15), kb = k0 + ((lane >> 4) * 8);
  const __bf16* p = W + (size_t)n * ldw + kb;
  const v8bf lo = *(const v8bf*)p;
  const v8bf hi = *(const v8bf*)(p + 16);
  v16bf b;
#pragma unroll
  for (int e = 0; e < 8; ++e) { b[e] = lo[e]; b[8 + e] = hi[e]; }
  return b;
}

__device__ inline float sigmoidf_(float x) { return 1.0f / (1.0f + __expf(-x)); }

// --------------------------- precompute kernels ----------------------------

// initial state: h only needed as bf16 (GEMM A); c needs f32 + bf16 copy
__global__ __launch_bounds__(256) void k_init(const float* __restrict__ h0,
                                              const float* __restrict__ c0,
                                              __bf16* __restrict__ hbf,
                                              float* __restrict__ cbuf,
                                              __bf16* __restrict__ cbf) {
  int i = blockIdx.x * 256 + threadIdx.x;
  if (i < kST) {
    hbf[i] = (__bf16)h0[i];
    cbuf[i] = c0[i];
    cbf[i] = (__bf16)c0[i];
  }
}

// generic f32 -> bf16 weight conversion
__global__ __launch_bounds__(256) void k_tobf(const float* __restrict__ s,
                                              __bf16* __restrict__ d, int n) {
  int i = blockIdx.x * 256 + threadIdx.x;
  if (i < n) d[i] = (__bf16)s[i];
}

// bf16 copy of w1[:, 512:]  (192 x 192)
__global__ __launch_bounds__(256) void k_w1x(const float* __restrict__ w1,
                                             __bf16* __restrict__ wx) {
  int idx = blockIdx.x * 256 + threadIdx.x;
  if (idx >= kN1 * kN1) return;
  int j = idx / kN1, k = idx % kN1;
  wx[idx] = (__bf16)w1[j * kKA + 512 + k];
}

// W12[j,k] = sum_q w2[j,q] * w1[q,k]   (192 x 512), bf16 (B operand)
__global__ __launch_bounds__(256) void k_w12(const float* __restrict__ w1,
                                             const float* __restrict__ w2,
                                             __bf16* __restrict__ w12) {
  int idx = blockIdx.x * 256 + threadIdx.x;
  if (idx >= kN1 * 512) return;
  int j = idx / 512, k = idx % 512;
  float s = 0.f;
  for (int q = 0; q < kN1; ++q) s += w2[j * kN1 + q] * w1[q * kKA + k];
  w12[idx] = (__bf16)s;
}

// c2vec[j] = b2[j] + sum_k w2[j,k]*b1[k]
__global__ __launch_bounds__(256) void k_c2vec(const float* __restrict__ w2,
                                               const float* __restrict__ b1,
                                               const float* __restrict__ b2,
                                               float* __restrict__ c2vec) {
  int j = blockIdx.x * 256 + threadIdx.x;
  if (j >= kN1) return;
  float s = b2[j];
  for (int k = 0; k < kN1; ++k) s += w2[j * kN1 + k] * b1[k];
  c2vec[j] = s;
}

// Xw1[(b,d),j] = sum_t input[b,t,d] * w1x[j,t]    M=32768, N=192, K=192
__global__ __launch_bounds__(256) void k_xw1(const float* __restrict__ inp,
                                             const __bf16* __restrict__ wx,
                                             __bf16* __restrict__ xw1) {
  const int lane = threadIdx.x & 31;
  const int wid = blockIdx.x * 8 + (threadIdx.x >> 5); // one 16x16 tile per wave
  const int mt = wid / 12, nt = wid % 12;
  const int m0 = mt * 16, n0 = nt * 16;
  v8f acc = {};
  for (int k0 = 0; k0 < kN1; k0 += 32) {
    v16bf a = load_a_frag_f32([&](int r, int kk) {
      const int m = m0 + r, b = m >> 7, d = m & 127, tt = k0 + kk;
      return inp[(b * kT + tt) * kD + d];
    }, lane);
    v16bf bb = load_b_frag_bf(wx, kN1, n0, k0, lane);
    acc = wmma_bf16(a, bb, acc);
  }
  const int n = n0 + (lane & 15);
#pragma unroll
  for (int v = 0; v < 8; ++v) {
    const int m = m0 + v + ((lane >> 4) * 8);
    xw1[m * kN1 + n] = (__bf16)acc[v];
  }
}

// P2 = Xw1 @ w2^T   M=32768, N=192, K=192  (f32 out, L2 resident)
__global__ __launch_bounds__(256) void k_p2(const __bf16* __restrict__ xw1,
                                            const __bf16* __restrict__ w2bf,
                                            float* __restrict__ p2) {
  const int lane = threadIdx.x & 31;
  const int wid = blockIdx.x * 8 + (threadIdx.x >> 5);
  const int mt = wid / 12, nt = wid % 12;
  const int m0 = mt * 16, n0 = nt * 16;
  v8f acc = {};
  for (int k0 = 0; k0 < kN1; k0 += 32) {
    v16bf a = load_a_frag_bf([&](int r, int k) {
      return *(const v2bf*)(xw1 + (size_t)(m0 + r) * kN1 + k0 + k);
    }, lane);
    v16bf bb = load_b_frag_bf(w2bf, kN1, n0, k0, lane);
    acc = wmma_bf16(a, bb, acc);
  }
  const int n = n0 + (lane & 15);
#pragma unroll
  for (int v = 0; v < 8; ++v) {
    const int m = m0 + v + ((lane >> 4) * 8);
    p2[m * kN1 + n] = acc[v];
  }
}

// ----------------------------- per-step kernels ----------------------------

// S1: hc2[(b*2+dir), j] = [h(dir,b)|c(dir,b)] @ W12^T + c2vec   M=512,K=512,N=192
__global__ __launch_bounds__(256) void k_hc(const __bf16* __restrict__ hbf,
                                            const __bf16* __restrict__ cbf,
                                            const __bf16* __restrict__ w12,
                                            const float* __restrict__ c2vec,
                                            float* __restrict__ hc2) {
  const int lane = threadIdx.x & 31;
  const int wid = blockIdx.x * 8 + (threadIdx.x >> 5); // 384 tiles
  const int mt = wid / 12, nt = wid % 12;
  const int m0 = mt * 16, n0 = nt * 16;
  v8f acc = {};
  for (int k0 = 0; k0 < 512; k0 += 32) {
    // whole fragment lies on one side of the h|c boundary (k0 multiple of 32)
    const __bf16* src = (k0 < kH) ? hbf : cbf;
    const int kbase = (k0 < kH) ? k0 : (k0 - kH);
    v16bf a = load_a_frag_bf([&](int r, int k) {
      const int m = m0 + r, b = m >> 1, dir = m & 1;
      return *(const v2bf*)(src + (size_t)dir * kB * kH + (size_t)b * kH + kbase + k);
    }, lane);
    v16bf bb = load_b_frag_bf(w12, 512, n0, k0, lane);
    acc = wmma_bf16(a, bb, acc);
  }
  const int n = n0 + (lane & 15);
  const float cv = c2vec[n];
#pragma unroll
  for (int v = 0; v < 8; ++v) {
    const int m = m0 + v + ((lane >> 4) * 8);
    hc2[m * kN1 + n] = acc[v] + cv;
  }
}

// S2 (fused): logits + softmax over D + xw. One block per batch row.
// 16 waves x 8 rows; hc2 rows (2 per b) and w3 staged in LDS.
__global__ __launch_bounds__(512) void k_attn(const float* __restrict__ hc2,
                                              const float* __restrict__ p2,
                                              const float* __restrict__ w3,
                                              const float* __restrict__ b3,
                                              const float* __restrict__ inp,
                                              float* __restrict__ out_w,
                                              __bf16* __restrict__ xwbf, int t) {
  __shared__ float hcsh[2 * kN1];
  __shared__ float w3sh[kN1];
  __shared__ float sm[kD];
  __shared__ float red[kD];
  const int tid = threadIdx.x, b = blockIdx.x;
  for (int j = tid; j < 2 * kN1; j += 512) hcsh[j] = hc2[(size_t)(b * 2) * kN1 + j];
  for (int j = tid; j < kN1; j += 512) w3sh[j] = w3[j];
  __syncthreads();

  const int lane = tid & 31, w = tid >> 5; // 16 waves
#pragma unroll
  for (int i = 0; i < 8; ++i) {
    const int d = w * 8 + i, dir = d & 1;
    const float* prow = p2 + ((size_t)b * kD + d) * kN1;
    __builtin_prefetch(prow, 0, 1);               // global_prefetch_b8
    float s = 0.f;
#pragma unroll
    for (int j = lane; j < kN1; j += 32)
      s += w3sh[j] * tanhf(hcsh[dir * kN1 + j] + prow[j]);
#pragma unroll
    for (int off = 16; off > 0; off >>= 1) s += __shfl_down(s, off, 32);
    if (lane == 0) sm[d] = s + b3[0];
  }
  __syncthreads();

  float v = 0.f;
  if (tid < kD) { v = sm[tid]; red[tid] = v; }
  __syncthreads();
  for (int s = 64; s > 0; s >>= 1) {
    if (tid < s) red[tid] = fmaxf(red[tid], red[tid + s]);
    __syncthreads();
  }
  const float mx = red[0];
  __syncthreads();
  float e = 0.f;
  if (tid < kD) { e = __expf(v - mx); red[tid] = e; }
  __syncthreads();
  for (int s = 64; s > 0; s >>= 1) {
    if (tid < s) red[tid] += red[tid + s];
    __syncthreads();
  }
  if (tid < kD) {
    const float attn = e / red[0];
    out_w[((size_t)b * kT + t) * kD + tid] = attn;
    xwbf[b * kD + tid] = (__bf16)(attn * inp[((size_t)b * kT + t) * kD + tid]);
  }
}

// S3: bidirectional LSTM cell. Wave owns a 16(batch)x16(hidden) tile and all 4
// gate accumulators -> state update stays in WMMA accumulator registers.
__global__ __launch_bounds__(256) void k_lstm(const __bf16* __restrict__ xwbf,
                                              const __bf16* __restrict__ hbf,
                                              const float* __restrict__ ccur,
                                              const __bf16* __restrict__ wihf,
                                              const __bf16* __restrict__ whhf,
                                              const float* __restrict__ bihf,
                                              const float* __restrict__ bhhf,
                                              const __bf16* __restrict__ wihb,
                                              const __bf16* __restrict__ whhb,
                                              const float* __restrict__ bihb,
                                              const float* __restrict__ bhhb,
                                              __bf16* __restrict__ hbfn,
                                              float* __restrict__ cnext,
                                              __bf16* __restrict__ cbfn,
                                              float* __restrict__ out_enc, int t) {
  const int lane = threadIdx.x & 31;
  const int wid = blockIdx.x * 8 + (threadIdx.x >> 5); // 512 waves
  const int dir = wid >> 8, mt = (wid >> 4) & 15, htile = wid & 15;
  const int m0 = mt * 16, n0 = htile * 16;
  const __bf16* wih = dir ? wihb : wihf;
  const __bf16* whh = dir ? whhb : whhf;
  const float* bih = dir ? bihb : bihf;
  const float* bhh = dir ? bhhb : bhhf;

  v8f acc[4] = {v8f{}, v8f{}, v8f{}, v8f{}};
  // x-part: gates += xw @ w_ih^T, K = D = 128
  for (int k0 = 0; k0 < kD; k0 += 32) {
    v16bf a = load_a_frag_bf([&](int r, int k) {
      return *(const v2bf*)(xwbf + (size_t)(m0 + r) * kD + k0 + k);
    }, lane);
#pragma unroll
    for (int q = 0; q < 4; ++q) {
      v16bf bb = load_b_frag_bf(wih, kD, q * kH + n0, k0, lane);
      acc[q] = wmma_bf16(a, bb, acc[q]);
    }
  }
  // h-part: gates += h @ w_hh^T, K = H = 256
  for (int k0 = 0; k0 < kH; k0 += 32) {
    v16bf a = load_a_frag_bf([&](int r, int k) {
      return *(const v2bf*)(hbf + (size_t)dir * kB * kH + (size_t)(m0 + r) * kH + k0 + k);
    }, lane);
#pragma unroll
    for (int q = 0; q < 4; ++q) {
      v16bf bb = load_b_frag_bf(whh, kH, q * kH + n0, k0, lane);
      acc[q] = wmma_bf16(a, bb, acc[q]);
    }
  }

  const int nn = n0 + (lane & 15);
  const float bI = bih[0 * kH + nn] + bhh[0 * kH + nn];
  const float bF = bih[1 * kH + nn] + bhh[1 * kH + nn];
  const float bG = bih[2 * kH + nn] + bhh[2 * kH + nn];
  const float bO = bih[3 * kH + nn] + bhh[3 * kH + nn];
#pragma unroll
  for (int v = 0; v < 8; ++v) {
    const int m = m0 + v + ((lane >> 4) * 8);
    const float gi = sigmoidf_(acc[0][v] + bI);
    const float gf = sigmoidf_(acc[1][v] + bF);
    const float gg = tanhf(acc[2][v] + bG);
    const float go = sigmoidf_(acc[3][v] + bO);
    const int sidx = dir * kB * kH + m * kH + nn;
    const float c2 = gf * ccur[sidx] + gi * gg;
    const float hv = go * tanhf(c2);
    cnext[sidx] = c2;
    cbfn[sidx] = (__bf16)c2;
    hbfn[sidx] = (__bf16)hv;
    // torch h2.view(B, -1) raw flat reshape of (2,B,H) -> (B, 2H)
    const int r = sidx >> 9, col = sidx & 511;
    out_enc[((size_t)r * kT + t) * (2 * kH) + col] = hv;
  }
}

// ------------------------------- host driver -------------------------------

extern "C" void kernel_launch(void* const* d_in, const int* in_sizes, int n_in,
                              void* d_out, int out_size, void* d_ws, size_t ws_size,
                              hipStream_t stream) {
  const float* inp   = (const float*)d_in[0];
  const float* h0    = (const float*)d_in[1];
  const float* c0    = (const float*)d_in[2];
  const float* w1    = (const float*)d_in[3];
  const float* b1    = (const float*)d_in[4];
  const float* w2    = (const float*)d_in[5];
  const float* b2    = (const float*)d_in[6];
  const float* w3    = (const float*)d_in[7];
  const float* b3    = (const float*)d_in[8];
  const float* wih_f = (const float*)d_in[9];
  const float* whh_f = (const float*)d_in[10];
  const float* bih_f = (const float*)d_in[11];
  const float* bhh_f = (const float*)d_in[12];
  const float* wih_b = (const float*)d_in[13];
  const float* whh_b = (const float*)d_in[14];
  const float* bih_b = (const float*)d_in[15];
  const float* bhh_b = (const float*)d_in[16];

  float* out_w   = (float*)d_out;                 // (B, T, D)
  float* out_enc = out_w + (size_t)kB * kT * kD;  // (B, T, 2H)

  // workspace carving (256B aligned)
  char* base = (char*)d_ws;
  size_t off = 0;
  auto carve = [&](size_t bytes) {
    char* p = base + off;
    off = (off + bytes + 255) & ~(size_t)255;
    return p;
  };
  __bf16* xw1    = (__bf16*)carve((size_t)kBD * kN1 * sizeof(__bf16)); // 12.6MB
  float*  p2     = (float*)carve((size_t)kBD * kN1 * sizeof(float));   // 25.2MB
  __bf16* w12    = (__bf16*)carve((size_t)kN1 * 512 * sizeof(__bf16));
  __bf16* wx     = (__bf16*)carve((size_t)kN1 * kN1 * sizeof(__bf16));
  __bf16* w2bf   = (__bf16*)carve((size_t)kN1 * kN1 * sizeof(__bf16));
  __bf16* wihfbf = (__bf16*)carve((size_t)4 * kH * kD * sizeof(__bf16));
  __bf16* whhfbf = (__bf16*)carve((size_t)4 * kH * kH * sizeof(__bf16));
  __bf16* wihbbf = (__bf16*)carve((size_t)4 * kH * kD * sizeof(__bf16));
  __bf16* whhbbf = (__bf16*)carve((size_t)4 * kH * kH * sizeof(__bf16));
  float*  c2vec  = (float*)carve(kN1 * sizeof(float));
  float*  hc2    = (float*)carve((size_t)512 * kN1 * sizeof(float));
  __bf16* xwbf   = (__bf16*)carve((size_t)kBD * sizeof(__bf16));
  __bf16* hbf0   = (__bf16*)carve((size_t)kST * sizeof(__bf16));
  __bf16* hbf1   = (__bf16*)carve((size_t)kST * sizeof(__bf16));
  float*  cbuf0  = (float*)carve((size_t)kST * sizeof(float));
  float*  cbuf1  = (float*)carve((size_t)kST * sizeof(float));
  __bf16* cbf0   = (__bf16*)carve((size_t)kST * sizeof(__bf16));
  __bf16* cbf1   = (__bf16*)carve((size_t)kST * sizeof(__bf16));
  (void)ws_size; (void)in_sizes; (void)n_in; (void)out_size;

  // ---- one-time (per call) precompute ----
  k_init<<<(kST + 255) / 256, 256, 0, stream>>>(h0, c0, hbf0, cbuf0, cbf0);
  k_w12<<<(kN1 * 512 + 255) / 256, 256, 0, stream>>>(w1, w2, w12);
  k_w1x<<<(kN1 * kN1 + 255) / 256, 256, 0, stream>>>(w1, wx);
  k_tobf<<<(kN1 * kN1 + 255) / 256, 256, 0, stream>>>(w2, w2bf, kN1 * kN1);
  k_tobf<<<(4 * kH * kD + 255) / 256, 256, 0, stream>>>(wih_f, wihfbf, 4 * kH * kD);
  k_tobf<<<(4 * kH * kH + 255) / 256, 256, 0, stream>>>(whh_f, whhfbf, 4 * kH * kH);
  k_tobf<<<(4 * kH * kD + 255) / 256, 256, 0, stream>>>(wih_b, wihbbf, 4 * kH * kD);
  k_tobf<<<(4 * kH * kH + 255) / 256, 256, 0, stream>>>(whh_b, whhbbf, 4 * kH * kH);
  k_c2vec<<<1, 256, 0, stream>>>(w2, b1, b2, c2vec);
  k_xw1<<<(kBD / 16) * 12 / 8, 256, 0, stream>>>(inp, wx, xw1);   // 3072 blocks
  k_p2<<<(kBD / 16) * 12 / 8, 256, 0, stream>>>(xw1, w2bf, p2);   // 3072 blocks

  // ---- 192 sequential steps (3 kernels/step) ----
  for (int t = 0; t < kT; ++t) {
    __bf16* hb = (t & 1) ? hbf1 : hbf0;
    __bf16* hn = (t & 1) ? hbf0 : hbf1;
    float*  cc = (t & 1) ? cbuf1 : cbuf0;
    float*  cn = (t & 1) ? cbuf0 : cbuf1;
    __bf16* cb = (t & 1) ? cbf1 : cbf0;
    __bf16* cbn = (t & 1) ? cbf0 : cbf1;
    k_hc<<<48, 256, 0, stream>>>(hb, cb, w12, c2vec, hc2);            // 384 tiles
    k_attn<<<kB, 512, 0, stream>>>(hc2, p2, w3, b3, inp, out_w, xwbf, t);
    k_lstm<<<64, 256, 0, stream>>>(xwbf, hb, cc,
                                   wihfbf, whhfbf, bih_f, bhh_f,
                                   wihbbf, whhbbf, bih_b, bhh_b,
                                   hn, cn, cbn, out_enc, t);          // 512 waves
  }
}
